// GATRouter_54563264528896
// MI455X (gfx1250) — compile-verified, matrix operand order checked by reading, not run on hardware
//
#include <hip/hip_runtime.h>

// ---------------- problem constants (from reference) ----------------
#define N_NODES 50000
#define N_EDGES 800000
#define ETOT    850000   // edges + self loops
#define HID     256
#define HEADS   4
#define CPH     64
#define NEG_SLOPE 0.2f
#define NB      8        // nodes per block in the MLP-head kernel

typedef __attribute__((ext_vector_type(16))) __bf16          v16bf;
typedef __attribute__((ext_vector_type(16))) unsigned short  v16u;
typedef __attribute__((ext_vector_type(8)))  float           v8f;

struct Q2 { uint4 lo, hi; };   // 32B container -> bit_cast to v16bf

// ---------------- small helpers ----------------
__device__ __forceinline__ unsigned short f2bf(float f) {
  unsigned u = __float_as_uint(f);
  return (unsigned short)((u + 0x7FFFu + ((u >> 16) & 1u)) >> 16); // RNE
}
__device__ __forceinline__ float bf2f(unsigned s) { return __uint_as_float(s << 16); }

// order-preserving float<->uint encoding for atomicMax-based segment max
__device__ __forceinline__ unsigned ordenc(float f) {
  unsigned u = __float_as_uint(f);
  return (u & 0x80000000u) ? ~u : (u | 0x80000000u);
}
__device__ __forceinline__ float orddec(unsigned x) {
  return (x & 0x80000000u) ? __uint_as_float(x & 0x7FFFFFFFu) : __uint_as_float(~x);
}

__device__ __forceinline__ void unpack_bf8(uint4 v, float* f) {
  f[0] = bf2f(v.x & 0xFFFFu); f[1] = bf2f(v.x >> 16);
  f[2] = bf2f(v.y & 0xFFFFu); f[3] = bf2f(v.y >> 16);
  f[4] = bf2f(v.z & 0xFFFFu); f[5] = bf2f(v.z >> 16);
  f[6] = bf2f(v.w & 0xFFFFu); f[7] = bf2f(v.w >> 16);
}

// ---------------- utility kernels ----------------
__global__ void zero_f32(float* __restrict__ p, int n) {
  int i = blockIdx.x * blockDim.x + threadIdx.x;
  if (i < n) p[i] = 0.0f;
}
__global__ void cvt_bf16(const float* __restrict__ src, unsigned short* __restrict__ dst, int n) {
  int i = blockIdx.x * blockDim.x + threadIdx.x;
  if (i < n) dst[i] = f2bf(src[i]);
}

// ---------------- WMMA GEMM:  Y[n][o] = sum_k X[n][k]*W[o][k] + bias[o] ----------------
// X: (nrows, KIN) bf16 row-major.  W: (256, KIN) bf16 row-major.  Y: (nrows, 256) bf16.
// 8 waves/block; each wave computes one 16x16 tile via v_wmma_f32_16x16x32_bf16,
// sharing the A row-tile through L0/L1.
template <int KIN>
__global__ void gemm_wmma(const unsigned short* __restrict__ X,
                          const unsigned short* __restrict__ W,
                          const float* __restrict__ bias, unsigned short* __restrict__ Y)
{
  const int lane  = threadIdx.x & 31;
  const int wv    = threadIdx.x >> 5;                // 0..7
  const int m0    = blockIdx.x * 16;                 // node-row tile
  const int o0    = (blockIdx.y * 8 + wv) * 16;      // output-col tile
  const int arow  = m0 + (lane & 15);
  const int wrow  = o0 + (lane & 15);                // B column == W row (column-major B frag)
  const int khalf = (lane >> 4) << 3;                // 0 for lanes 0-15, 8 for lanes 16-31

  v8f acc = {};
  for (int ks = 0; ks < KIN; ks += 32) {
    const unsigned short* xp = X + (size_t)arow * KIN + ks + khalf;
    const unsigned short* wp = W + (size_t)wrow * KIN + ks + khalf;
    Q2 fa, fb;
    fa.lo = *reinterpret_cast<const uint4*>(xp);     // K = ks+khalf .. +7
    fb.lo = *reinterpret_cast<const uint4*>(wp);
    if ((KIN % 32) == 0) {                           // upper K half in-range (KIN=256)
      fa.hi = *reinterpret_cast<const uint4*>(xp + 16);
      fb.hi = *reinterpret_cast<const uint4*>(wp + 16);
    } else {                                         // KIN=16: zero-pad K 16..31
      fa.hi = make_uint4(0u, 0u, 0u, 0u);
      fb.hi = make_uint4(0u, 0u, 0u, 0u);
    }
    acc = __builtin_amdgcn_wmma_f32_16x16x32_bf16(false, __builtin_bit_cast(v16bf, fa),
                                                  false, __builtin_bit_cast(v16bf, fb),
                                                  (short)0, acc, false, false);
  }

  // D layout: VGPR r -> row m0+r (lanes 0-15) / m0+8+r (lanes 16-31), col o0+(lane&15)
  const int col   = o0 + (lane & 15);
  const int rbase = m0 + ((lane >= 16) ? 8 : 0);
  const float bv  = bias[col];
#pragma unroll
  for (int r = 0; r < 8; ++r)
    Y[(size_t)(rbase + r) * HID + col] = f2bf(acc[r] + bv);
}

// ---------------- edge pass 1: per-edge attention logits + segment max ----------------
// one wave per edge; lanes 0-7 -> head0, 8-15 -> head1, ... ; 8 channels per lane
__global__ void edge_logits(const unsigned short* __restrict__ XL,
                            const unsigned short* __restrict__ XR,
                            const float* __restrict__ att,
                            const int* __restrict__ ei,
                            float* __restrict__ eBuf,
                            unsigned* __restrict__ emaxU)
{
  const int wave = blockIdx.x * (blockDim.x >> 5) + (threadIdx.x >> 5);
  if (wave >= ETOT) return;
  const int lane = threadIdx.x & 31;
  int src, dst;
  if (wave < N_EDGES) { src = ei[wave]; dst = ei[N_EDGES + wave]; }
  else                { src = dst = wave - N_EDGES; }

  const int head = lane >> 3, sub = lane & 7;
  const int ch   = head * CPH + sub * 8;

  uint4 av = *reinterpret_cast<const uint4*>(XL + (size_t)src * HID + ch);
  uint4 bv = *reinterpret_cast<const uint4*>(XR + (size_t)dst * HID + ch);
  float xs[8], xr[8];
  unpack_bf8(av, xs); unpack_bf8(bv, xr);
  const float4* ap = reinterpret_cast<const float4*>(att + ch);
  float4 w0 = ap[0], w1 = ap[1];
  const float wv[8] = {w0.x, w0.y, w0.z, w0.w, w1.x, w1.y, w1.z, w1.w};

  float s = 0.0f;
#pragma unroll
  for (int j = 0; j < 8; ++j) {
    float m = xs[j] + xr[j];
    m = (m > 0.0f) ? m : NEG_SLOPE * m;       // leaky_relu
    s += wv[j] * m;
  }
  s += __shfl_xor(s, 1, 32);
  s += __shfl_xor(s, 2, 32);
  s += __shfl_xor(s, 4, 32);                  // reduce 8-lane head group
  if (sub == 0) {
    eBuf[(size_t)wave * HEADS + head] = s;
    atomicMax(&emaxU[dst * HEADS + head], ordenc(s));
  }
}

// ---------------- edge pass 2: exp(e - max) and denominator ----------------
__global__ void edge_softmax_den(float* __restrict__ eBuf,
                                 const unsigned* __restrict__ emaxU,
                                 const int* __restrict__ ei,
                                 float* __restrict__ den)
{
  const int idx = blockIdx.x * blockDim.x + threadIdx.x;   // over ETOT*HEADS
  if (idx >= ETOT * HEADS) return;
  const int e = idx >> 2, h = idx & 3;
  const int dst = (e < N_EDGES) ? ei[N_EDGES + e] : (e - N_EDGES);
  const float ee = __expf(eBuf[idx] - orddec(emaxU[dst * HEADS + h]));
  eBuf[idx] = ee;
  unsafeAtomicAdd(&den[dst * HEADS + h], ee);
}

// ---------------- edge pass 3: alpha-weighted scatter of xl[src] ----------------
__global__ void edge_scatter(const unsigned short* __restrict__ XL,
                             const float* __restrict__ eBuf,
                             const float* __restrict__ den,
                             const int* __restrict__ ei,
                             float* __restrict__ ACC)
{
  const int wave = blockIdx.x * (blockDim.x >> 5) + (threadIdx.x >> 5);
  if (wave >= ETOT) return;
  const int lane = threadIdx.x & 31;
  int src, dst;
  if (wave < N_EDGES) { src = ei[wave]; dst = ei[N_EDGES + wave]; }
  else                { src = dst = wave - N_EDGES; }

  const int head = lane >> 3, sub = lane & 7;
  const int ch   = head * CPH + sub * 8;
  const float alpha = eBuf[(size_t)wave * HEADS + head] /
                      (den[dst * HEADS + head] + 1e-16f);

  uint4 av = *reinterpret_cast<const uint4*>(XL + (size_t)src * HID + ch);
  float xs[8]; unpack_bf8(av, xs);
  float* op = ACC + (size_t)dst * HID + ch;
#pragma unroll
  for (int j = 0; j < 8; ++j)
    unsafeAtomicAdd(op + j, alpha * xs[j]);
}

// ---------------- fused conv-bias + LayerNorm + ELU -> bf16 hidden ----------------
__global__ void ln_elu(const float* __restrict__ ACC, const float* __restrict__ cb,
                       const float* __restrict__ g, const float* __restrict__ b,
                       unsigned short* __restrict__ Hb)
{
  __shared__ float red[HID];
  const int n = blockIdx.x, t = threadIdx.x;
  const float v = ACC[(size_t)n * HID + t] + cb[t];
  red[t] = v; __syncthreads();
  for (int s = 128; s > 0; s >>= 1) { if (t < s) red[t] += red[t + s]; __syncthreads(); }
  const float mean = red[0] * (1.0f / HID);
  __syncthreads();
  const float d = v - mean;
  red[t] = d * d; __syncthreads();
  for (int s = 128; s > 0; s >>= 1) { if (t < s) red[t] += red[t + s]; __syncthreads(); }
  const float var = red[0] * (1.0f / HID);
  const float y = d * rsqrtf(var + 1e-5f) * g[t] + b[t];
  Hb[(size_t)n * HID + t] = f2bf((y > 0.0f) ? y : (__expf(y) - 1.0f));
}

// ---------------- actor/critic MLP heads + mean state value ----------------
// 64 threads, NB nodes per block: weight rows stream once per block, h rows staged in LDS.
__global__ void mlp_heads(const unsigned short* __restrict__ Hb,
                          const float* __restrict__ aW1, const float* __restrict__ ab1,
                          const float* __restrict__ aW2, const float* __restrict__ ab2,
                          const float* __restrict__ vW1, const float* __restrict__ vb1,
                          const float* __restrict__ vW2, const float* __restrict__ vb2,
                          float* __restrict__ out)
{
  __shared__ float hs[NB * HID];
  __shared__ float ra[64 * NB], rv[64 * NB];
  const int n0 = blockIdx.x * NB;
  const int t  = threadIdx.x;                       // 0..63
  for (int i = t; i < NB * HID; i += 64) hs[i] = bf2f(Hb[(size_t)n0 * HID + i]);
  __syncthreads();

  float sa[NB], sv[NB];
#pragma unroll
  for (int nn = 0; nn < NB; ++nn) { sa[nn] = ab1[t]; sv[nn] = vb1[t]; }
  const float* wa  = aW1 + t * HID;
  const float* wvp = vW1 + t * HID;
  for (int c = 0; c < HID; ++c) {
    const float a = wa[c], v = wvp[c];
#pragma unroll
    for (int nn = 0; nn < NB; ++nn) {
      const float h = hs[nn * HID + c];
      sa[nn] += h * a; sv[nn] += h * v;
    }
  }
  const float a2 = aW2[t], v2 = vW2[t];
#pragma unroll
  for (int nn = 0; nn < NB; ++nn) {
    ra[t * NB + nn] = fmaxf(sa[nn], 0.0f) * a2;
    rv[t * NB + nn] = fmaxf(sv[nn], 0.0f) * v2;
  }
  __syncthreads();
  for (int s = 32; s > 0; s >>= 1) {
    if (t < s) {
#pragma unroll
      for (int nn = 0; nn < NB; ++nn) {
        ra[t * NB + nn] += ra[(t + s) * NB + nn];
        rv[t * NB + nn] += rv[(t + s) * NB + nn];
      }
    }
    __syncthreads();
  }
  if (t < NB) out[n0 + t] = ra[t] + ab2[0];
  if (t == 0) {
    float s = 0.0f;
#pragma unroll
    for (int nn = 0; nn < NB; ++nn) s += rv[nn];
    unsafeAtomicAdd(out + N_NODES, (s + NB * vb2[0]) * (1.0f / N_NODES));
  }
}

// ---------------- host orchestration ----------------
extern "C" void kernel_launch(void* const* d_in, const int* in_sizes, int n_in,
                              void* d_out, int out_size, void* d_ws, size_t ws_size,
                              hipStream_t stream) {
  (void)in_sizes; (void)n_in; (void)out_size; (void)ws_size;
  const float* x   = (const float*)d_in[0];
  const int*   ei  = (const int*)  d_in[1];
  const float* Wl0 = (const float*)d_in[2],  *bl0 = (const float*)d_in[3];
  const float* Wr0 = (const float*)d_in[4],  *br0 = (const float*)d_in[5];
  const float* att0= (const float*)d_in[6],  *cb0 = (const float*)d_in[7];
  const float* g0  = (const float*)d_in[8],  *b0  = (const float*)d_in[9];
  const float* Wl1 = (const float*)d_in[10], *bl1 = (const float*)d_in[11];
  const float* Wr1 = (const float*)d_in[12], *br1 = (const float*)d_in[13];
  const float* att1= (const float*)d_in[14], *cb1 = (const float*)d_in[15];
  const float* g1  = (const float*)d_in[16], *b1  = (const float*)d_in[17];
  const float* aW1 = (const float*)d_in[18], *ab1 = (const float*)d_in[19];
  const float* aW2 = (const float*)d_in[20], *ab2 = (const float*)d_in[21];
  const float* vW1 = (const float*)d_in[22], *vb1 = (const float*)d_in[23];
  const float* vW2 = (const float*)d_in[24], *vb2 = (const float*)d_in[25];
  float* out = (float*)d_out;

  // workspace layout (bytes); ~145 MB total, hot tables L2-resident
  char* ws = (char*)d_ws;
  unsigned short* XL   = (unsigned short*)(ws + 0);            // N*256 bf16 = 25.6 MB
  unsigned short* XR   = (unsigned short*)(ws + 25600000);     // 25.6 MB
  float*          ACC  = (float*)(ws + 51200000);              // N*256 f32 = 51.2 MB
  unsigned short* Hb   = (unsigned short*)(ws + 102400000);    // N*256 bf16 = 25.6 MB
  float*          EB   = (float*)(ws + 128000000);             // ETOT*4 f32 = 13.6 MB
  unsigned*       EMAX = (unsigned*)(ws + 141600000);          // N*4 = 0.8 MB
  float*          DEN  = (float*)(ws + 142400000);             // 0.8 MB
  unsigned short* X0b  = (unsigned short*)(ws + 143200000);    // N*16 bf16 = 1.6 MB
  unsigned short* Wl0b = (unsigned short*)(ws + 144800000);    // 256*16 bf16
  unsigned short* Wr0b = (unsigned short*)(ws + 144816384);
  unsigned short* Wl1b = (unsigned short*)(ws + 144832768);    // 256*256 bf16
  unsigned short* Wr1b = (unsigned short*)(ws + 144963840);

  const dim3 gemmGrid(N_NODES / 16, 2);                  // 8 waves/block -> 16 col tiles
  const int edgeBlocks = (ETOT + 7) / 8;                 // 8 waves/block, 1 wave/edge
  const int eh = ETOT * HEADS;
  auto Z = [&](float* p, int n) { zero_f32<<<(n + 255) / 256, 256, 0, stream>>>(p, n); };
  auto C = [&](const float* s, unsigned short* d, int n) {
    cvt_bf16<<<(n + 255) / 256, 256, 0, stream>>>(s, d, n);
  };

  // -------- one-time bf16 staging (per launch, deterministic) --------
  C(x,   X0b,  N_NODES * 16);
  C(Wl0, Wl0b, HID * 16);   C(Wr0, Wr0b, HID * 16);
  C(Wl1, Wl1b, HID * HID);  C(Wr1, Wr1b, HID * HID);

  // -------- layer 0 (K = 16) --------
  gemm_wmma<16><<<gemmGrid, 256, 0, stream>>>(X0b, Wl0b, bl0, XL);
  gemm_wmma<16><<<gemmGrid, 256, 0, stream>>>(X0b, Wr0b, br0, XR);
  Z(ACC, N_NODES * HID); Z((float*)EMAX, N_NODES * HEADS); Z(DEN, N_NODES * HEADS);
  edge_logits     <<<edgeBlocks, 256, 0, stream>>>(XL, XR, att0, ei, EB, EMAX);
  edge_softmax_den<<<(eh + 255) / 256, 256, 0, stream>>>(EB, EMAX, ei, DEN);
  edge_scatter    <<<edgeBlocks, 256, 0, stream>>>(XL, EB, DEN, ei, ACC);
  ln_elu          <<<N_NODES, HID, 0, stream>>>(ACC, cb0, g0, b0, Hb);

  // -------- layer 1 (K = 256) --------
  gemm_wmma<256><<<gemmGrid, 256, 0, stream>>>(Hb, Wl1b, bl1, XL);
  gemm_wmma<256><<<gemmGrid, 256, 0, stream>>>(Hb, Wr1b, br1, XR);
  Z(ACC, N_NODES * HID); Z((float*)EMAX, N_NODES * HEADS); Z(DEN, N_NODES * HEADS);
  edge_logits     <<<edgeBlocks, 256, 0, stream>>>(XL, XR, att1, ei, EB, EMAX);
  edge_softmax_den<<<(eh + 255) / 256, 256, 0, stream>>>(EB, EMAX, ei, DEN);
  edge_scatter    <<<edgeBlocks, 256, 0, stream>>>(XL, EB, DEN, ei, ACC);
  ln_elu          <<<N_NODES, HID, 0, stream>>>(ACC, cb1, g1, b1, Hb);  // h2 overwrites h1

  // -------- heads --------
  Z(out + N_NODES, 1);                                   // state_value accumulator
  mlp_heads<<<N_NODES / NB, 64, 0, stream>>>(Hb, aW1, ab1, aW2, ab2, vW1, vb1, vW2, vb2, out);
}
